// FeatureVGG_74010876444936
// MI455X (gfx1250) — matrix operand order repaired and checked by reading, not compiled, not run on hardware
//
#include <hip/hip_runtime.h>
#include <hip/hip_bf16.h>

typedef __attribute__((ext_vector_type(16))) _Float16 v16h;
typedef __attribute__((ext_vector_type(8)))  _Float16 v8h;
typedef __attribute__((ext_vector_type(8)))  float    v8f;

#define BLK_M 128
#define BLK_N 64
#define KCHUNK 32

// ---------------------------------------------------------------------------
// Weight prep: f32 OIHW -> f16 [tap][ci_chunk][co][32 ci]
// Each (tap, ci_chunk, co-block-of-64) B-tile is a contiguous 4KB block,
// ready for async global->LDS DMA.
// ---------------------------------------------------------------------------
__global__ void prep_weights(const float* __restrict__ w, _Float16* __restrict__ wt,
                             int Cin, int Cout) {
    int idx = blockIdx.x * blockDim.x + threadIdx.x;
    int total = Cout * Cin * 9;
    if (idx >= total) return;
    int tap = idx % 9;
    int ci  = (idx / 9) % Cin;
    int co  = idx / (9 * Cin);
    size_t dst = (((size_t)tap * (Cin >> 5) + (ci >> 5)) * Cout + co) * 32 + (ci & 31);
    wt[dst] = (_Float16)w[idx];
}

// ---------------------------------------------------------------------------
// Layer 0: direct conv 3->64 (tiny K, stays on VALU), f32 in -> f16 out + ReLU
// ---------------------------------------------------------------------------
__global__ void conv0_direct(const float* __restrict__ x, const float* __restrict__ w,
                             const float* __restrict__ b, _Float16* __restrict__ out,
                             int H, int W) {
    int idx = blockIdx.x * blockDim.x + threadIdx.x;
    int total = 8 * 64 * H * W;
    if (idx >= total) return;
    int xc = idx % W;
    int y  = (idx / W) % H;
    int co = (idx / (W * H)) % 64;
    int n  = idx / (64 * W * H);
    float acc = b[co];
    for (int ci = 0; ci < 3; ++ci) {
        for (int dy = -1; dy <= 1; ++dy) {
            int sy = y + dy;
            if (sy < 0 || sy >= H) continue;
            for (int dx = -1; dx <= 1; ++dx) {
                int sx = xc + dx;
                if (sx < 0 || sx >= W) continue;
                float iv = x[(((size_t)n * 3 + ci) * H + sy) * W + sx];
                float wv = w[(((size_t)co * 3 + ci) * 3 + (dy + 1)) * 3 + (dx + 1)];
                acc = fmaf(iv, wv, acc);
            }
        }
    }
    acc = acc > 0.f ? acc : 0.f;
    out[(((size_t)n * 64 + co) * H + y) * W + xc] = (_Float16)acc;
}

// ---------------------------------------------------------------------------
// Implicit-GEMM conv3x3 SAME via V_WMMA_F32_16X16X32_F16.
//   Block: 256 threads (8 wave32) -> 128(M) x 64(N) C tile
//   Wave:  32x32 -> 2 A-frags x 2 B-frags -> 4 WMMA accumulators (reg reuse)
//   B tile (weights) fetched with global_load_async_to_lds_b128 (ASYNCcnt)
// ---------------------------------------------------------------------------
__global__ __launch_bounds__(256)
void conv_wmma(const _Float16* __restrict__ act, const _Float16* __restrict__ wt,
               const float* __restrict__ bias, _Float16* __restrict__ out,
               int Cin, int Cout, int H, int W) {
    __shared__ _Float16 As[BLK_M * KCHUNK];  // 8KB [m][k], row = 64B
    __shared__ _Float16 Bs[BLK_N * KCHUNK];  // 4KB [n][k], row = 64B

    const int t    = threadIdx.x;
    const int wv   = t >> 5;
    const int lane = t & 31;
    const int m_sub = wv & 3;    // 4 sub-tiles of 32 rows -> 128
    const int n_sub = wv >> 2;   // 2 sub-tiles of 32 cols -> 64
    const int lm = lane & 15;
    const int kh = lane >> 4;

    const int batch = blockIdx.z;
    const int m0 = blockIdx.x * BLK_M;
    const int n0 = blockIdx.y * BLK_N;
    const int HW = H * W;
    const int nkc = Cin >> 5;   // K chunks per tap

    // A cooperative load: thread -> (spatial m, 16 channels)
    const int a_m  = t & 127;
    const int a_k0 = (t >> 7) << 4;   // 0 or 16
    const int gm = m0 + a_m;
    const int oy = gm / W;
    const int ox = gm % W;

    // async B DMA: each of 256 threads moves 16 contiguous bytes (4KB tile)
    const unsigned ldsB  = (unsigned)(size_t)(&Bs[0]) + (unsigned)t * 16u;
    const unsigned bvoff = (unsigned)t * 16u;

    v8f c00 = {}, c01 = {}, c10 = {}, c11 = {};

    for (int tap = 0; tap < 9; ++tap) {
        const int dy = tap / 3 - 1;
        const int dx = tap % 3 - 1;
        const int sy = oy + dy;
        const int sx = ox + dx;
        const bool inb = (sy >= 0) & (sy < H) & (sx >= 0) & (sx < W);
        const _Float16* aBase = act + (size_t)batch * Cin * HW + (size_t)sy * W + sx;
        const _Float16* bTap  = wt + ((size_t)tap * nkc * Cout + n0) * KCHUNK;

        for (int kc = 0; kc < nkc; ++kc) {
            __syncthreads();

            // kick off async weight-tile DMA (global -> LDS, ASYNCcnt)
            const char* bSrc = (const char*)(bTap + (size_t)kc * Cout * KCHUNK);
            asm volatile("global_load_async_to_lds_b128 %0, %1, %2"
                         :: "v"(ldsB), "v"(bvoff), "s"(bSrc) : "memory");

            // A tile gather (NCHW channel stride) overlapped with the DMA
#pragma unroll
            for (int e = 0; e < 16; ++e) {
                int k = a_k0 + e;
                _Float16 v = (_Float16)0.f;
                if (inb) v = aBase[(size_t)(kc * KCHUNK + k) * HW];
                As[a_m * KCHUNK + k] = v;
            }

            asm volatile("s_wait_asynccnt 0x0" ::: "memory");
            __syncthreads();

            // A fragments: lane holds K in [16*kh,16*kh+16)
            v16h a0, a1;
            {
                const _Float16* ar = &As[(m_sub * 32 + lm) * KCHUNK];
                v8h lo = *(const v8h*)(ar + 16 * kh);
                v8h hi = *(const v8h*)(ar + 16 * kh + 8);
#pragma unroll
                for (int i = 0; i < 8; ++i) { a0[i] = lo[i]; a0[i + 8] = hi[i]; }
            }
            {
                const _Float16* ar = &As[(m_sub * 32 + 16 + lm) * KCHUNK];
                v8h lo = *(const v8h*)(ar + 16 * kh);
                v8h hi = *(const v8h*)(ar + 16 * kh + 8);
#pragma unroll
                for (int i = 0; i < 8; ++i) { a1[i] = lo[i]; a1[i + 8] = hi[i]; }
            }
            // B fragments: lane = column, contiguous K run of 16
            const v16h b0 = *(const v16h*)(&Bs[(n_sub * 32 + lm) * KCHUNK] + 16 * kh);
            const v16h b1 = *(const v16h*)(&Bs[(n_sub * 32 + 16 + lm) * KCHUNK] + 16 * kh);

            c00 = __builtin_amdgcn_wmma_f32_16x16x32_f16(false, a0, false, b0,
                                                         (short)0, c00, false, false);
            c01 = __builtin_amdgcn_wmma_f32_16x16x32_f16(false, a0, false, b1,
                                                         (short)0, c01, false, false);
            c10 = __builtin_amdgcn_wmma_f32_16x16x32_f16(false, a1, false, b0,
                                                         (short)0, c10, false, false);
            c11 = __builtin_amdgcn_wmma_f32_16x16x32_f16(false, a1, false, b1,
                                                         (short)0, c11, false, false);
        }
    }

    // C/D layout: VGPR r -> M = r + 8*kh, N = lane&15.  Fuse bias + ReLU.
    const int on0 = n0 + n_sub * 32 + lm;
    const int on1 = on0 + 16;
    const float bi0 = bias[on0];
    const float bi1 = bias[on1];
    const size_t obatch = (size_t)batch * Cout * HW;
#pragma unroll
    for (int am = 0; am < 2; ++am) {
        const v8f d0 = am ? c10 : c00;
        const v8f d1 = am ? c11 : c01;
        const int mb = m0 + m_sub * 32 + am * 16 + 8 * kh;
#pragma unroll
        for (int r = 0; r < 8; ++r) {
            int m = mb + r;
            int y = m / W;
            int x = m % W;
            float v0 = d0[r] + bi0; v0 = v0 > 0.f ? v0 : 0.f;
            float v1 = d1[r] + bi1; v1 = v1 > 0.f ? v1 : 0.f;
            size_t sp = (size_t)y * W + x;
            out[obatch + (size_t)on0 * HW + sp] = (_Float16)v0;
            out[obatch + (size_t)on1 * HW + sp] = (_Float16)v1;
        }
    }
}

// ---------------------------------------------------------------------------
// 2x2 max pool, stride 2 (f16)
// ---------------------------------------------------------------------------
__global__ void maxpool2(const _Float16* __restrict__ in, _Float16* __restrict__ out,
                         int C, int H, int W) {
    int Ho = H >> 1, Wo = W >> 1;
    int idx = blockIdx.x * blockDim.x + threadIdx.x;
    int total = 8 * C * Ho * Wo;
    if (idx >= total) return;
    int x = idx % Wo;
    int y = (idx / Wo) % Ho;
    int c = (idx / (Wo * Ho)) % C;
    int n = idx / (C * Wo * Ho);
    const _Float16* p = in + (((size_t)n * C + c) * H + 2 * y) * W + 2 * x;
    float a = (float)p[0];
    float b = (float)p[1];
    float d = (float)p[W];
    float e = (float)p[W + 1];
    float m = fmaxf(fmaxf(a, b), fmaxf(d, e));
    out[(((size_t)n * C + c) * Ho + y) * Wo + x] = (_Float16)m;
}

// ---------------------------------------------------------------------------
// Bilinear point sampling, replicating the reference's exact weight rules.
// out[b][p][c] (already the transpose(0,2,1) layout), f32.
// ---------------------------------------------------------------------------
__global__ void sample_feats(const _Float16* __restrict__ feat, const float* __restrict__ points,
                             const int* __restrict__ mask, float* __restrict__ out,
                             int C, int H, int W) {
    int idx = blockIdx.x * blockDim.x + threadIdx.x;
    int total = 8 * 32 * C;
    if (idx >= total) return;
    int c = idx % C;
    int p = (idx / C) % 32;
    int b = idx / (32 * C);

    int ns = 0;
    for (int i = 0; i < 32; ++i) ns += mask[b * 32 + i];

    float px = points[((size_t)b * 32 + p) * 2 + 0];
    float py = points[((size_t)b * 32 + p) * 2 + 1];
    float step = 256.0f / (float)H;
    float xx = (px - step * 0.5f) * (float)H / 256.0f;
    float yy = (py - step * 0.5f) * (float)W / 256.0f;
    float x0f = floorf(xx);
    float y0f = floorf(yy);
    int x0 = (int)fminf(fmaxf(x0f, 0.f), (float)(W - 1));
    int x1 = (int)fminf(fmaxf(x0f + 1.f, 0.f), (float)(W - 1));
    int y0 = (int)fminf(fmaxf(y0f, 0.f), (float)(H - 1));
    int y1 = (int)fminf(fmaxf(y0f + 1.f, 0.f), (float)(H - 1));

    const _Float16* im = feat + ((size_t)b * C + c) * H * W;
    float Ia = (float)im[(size_t)y0 * W + x0];
    float Ib = (float)im[(size_t)y1 * W + x0];
    float Ic = (float)im[(size_t)y0 * W + x1];
    float Id = (float)im[(size_t)y1 * W + x1];

    bool xe = (x0 == x1), ye = (y0 == y1);
    float x0w = (xe && x0 == 0) ? (float)(x0 - 1) : (float)x0;
    float x1w = (xe && x0 != 0) ? (float)(x1 + 1) : (float)x1;
    float y0w = (ye && y0 == 0) ? (float)(y0 - 1) : (float)y0;
    float y1w = (ye && y0 != 0) ? (float)(y1 + 1) : (float)y1;

    float wa = (x1w - xx) * (y1w - yy);
    float wb = (x1w - xx) * (yy - y0w);
    float wc = (xx - x0w) * (y1w - yy);
    float wd = (xx - x0w) * (yy - y0w);
    float val = Ia * wa + Ib * wb + Ic * wc + Id * wd;
    float valid = (p < ns) ? 1.f : 0.f;
    out[idx] = val * valid;
}

// ---------------------------------------------------------------------------
// Host orchestration
// ---------------------------------------------------------------------------
static const int kCIN[11]  = {3, 64, 64, 128, 128, 256, 256, 256, 512, 512, 512};
static const int kCOUT[11] = {64, 64, 128, 128, 256, 256, 256, 512, 512, 512, 512};

extern "C" void kernel_launch(void* const* d_in, const int* in_sizes, int n_in,
                              void* d_out, int out_size, void* d_ws, size_t ws_size,
                              hipStream_t stream) {
    (void)in_sizes; (void)n_in; (void)out_size; (void)ws_size;
    const float* x      = (const float*)d_in[0];
    const float* points = (const float*)d_in[1];
    const int*   mask   = (const int*)d_in[2];
    const float* Wf[11];
    const float* Bf[11];
    for (int i = 0; i < 11; ++i) {
        Wf[i] = (const float*)d_in[3 + 2 * i];
        Bf[i] = (const float*)d_in[4 + 2 * i];
    }

    char* ws = (char*)d_ws;
    size_t off = 0;
    _Float16* wt[11] = {};
    for (int i = 1; i < 11; ++i) {
        wt[i] = (_Float16*)(ws + off);
        off += (size_t)kCIN[i] * kCOUT[i] * 9 * sizeof(_Float16);
        off = (off + 255) & ~(size_t)255;
    }
    const size_t maxAct = (size_t)8 * 64 * 256 * 256 * sizeof(_Float16);  // 67 MB
    _Float16* actA = (_Float16*)(ws + off); off += maxAct;
    _Float16* actB = (_Float16*)(ws + off); off += maxAct;
    _Float16* actF = (_Float16*)(ws + off); off += (size_t)8 * 512 * 32 * 32 * sizeof(_Float16);

    // weight transform (layers 1..10)
    for (int i = 1; i < 11; ++i) {
        int total = kCOUT[i] * kCIN[i] * 9;
        prep_weights<<<(total + 255) / 256, 256, 0, stream>>>(Wf[i], wt[i], kCIN[i], kCOUT[i]);
    }

    // layer 0 direct
    {
        int total = 8 * 64 * 256 * 256;
        conv0_direct<<<(total + 255) / 256, 256, 0, stream>>>(x, Wf[0], Bf[0], actA, 256, 256);
    }

    auto conv = [&](const _Float16* in, int li, _Float16* outp, int H) {
        dim3 grid((unsigned)(H * H / BLK_M), (unsigned)(kCOUT[li] / BLK_N), 8u);
        conv_wmma<<<grid, 256, 0, stream>>>(in, wt[li], Bf[li], outp, kCIN[li], kCOUT[li], H, H);
    };
    auto pool = [&](const _Float16* in, _Float16* outp, int C, int H) {
        int total = 8 * C * (H / 2) * (H / 2);
        maxpool2<<<(total + 255) / 256, 256, 0, stream>>>(in, outp, C, H, H);
    };

    conv(actA, 1, actB, 256);
    pool(actB, actA, 64, 256);        // -> 128
    conv(actA, 2, actB, 128);
    conv(actB, 3, actA, 128);
    pool(actA, actB, 128, 128);       // -> 64
    conv(actB, 4, actA, 64);
    conv(actA, 5, actB, 64);
    conv(actB, 6, actA, 64);
    pool(actA, actB, 256, 64);        // -> 32
    conv(actB, 7, actA, 32);
    conv(actA, 8, actF, 32);          // F feature map
    conv(actF, 9, actB, 32);
    pool(actB, actA, 512, 32);        // -> 16
    conv(actA, 10, actB, 16);         // U feature map

    float* outF = (float*)d_out;
    float* outU = outF + (size_t)8 * 32 * 512;
    int totalS = 8 * 32 * 512;
    sample_feats<<<(totalS + 255) / 256, 256, 0, stream>>>(actF, points, mask, outF, 512, 32, 32);
    sample_feats<<<(totalS + 255) / 256, 256, 0, stream>>>(actB, points, mask, outU, 512, 16, 16);
}
